// RecurrentRGCN_47064251629968
// MI455X (gfx1250) — matrix operand order repaired and checked by reading, not compiled, not run on hardware
//
#include <hip/hip_runtime.h>
#include <hip/hip_bf16.h>

// ---------------------------------------------------------------------------
// Problem constants (match reference)
// ---------------------------------------------------------------------------
#define NENT  40000
#define HD    200
#define R2    500
#define TT    8
#define EE    200000
#define MM    100000
#define RRELU_SLOPE ((1.0f/8.0f + 1.0f/3.0f) * 0.5f)

typedef __attribute__((ext_vector_type(16))) __bf16 v16bf;
typedef __attribute__((ext_vector_type(8)))  float  v8f;

union FragU { v16bf v; unsigned int u[8]; };

__device__ __forceinline__ unsigned short f32_to_bf16(float f) {
  unsigned int u = __float_as_uint(f);
  u += 0x7FFFu + ((u >> 16) & 1u);           // round-to-nearest-even
  return (unsigned short)(u >> 16);
}
__device__ __forceinline__ unsigned int pack_bf16x2(float lo, float hi) {
  return (unsigned int)f32_to_bf16(lo) | ((unsigned int)f32_to_bf16(hi) << 16);
}
__device__ __forceinline__ float sigmoidf_(float x) { return 1.0f / (1.0f + __expf(-x)); }

// ---------------------------------------------------------------------------
// Weight pre-pass: f32 [N,K] (or transpose of [K,N]) -> bf16 [Npad,Kpad],
// zero padded.  Run once per launch; all GEMMs then see a uniform bf16 [N,K]
// row-major B with 16B-aligned, async-copyable rows.
// ---------------------------------------------------------------------------
__global__ void k_conv_w(const float* __restrict__ W, unsigned short* __restrict__ out,
                         int N, int K, int Kpad, long total, int trans) {
  long i = (long)blockIdx.x * blockDim.x + threadIdx.x;
  if (i >= total) return;
  int n = (int)(i / Kpad), k = (int)(i % Kpad);
  float v = 0.0f;
  if (n < N && k < K) v = trans ? W[(long)k * N + n] : W[(long)n * K + k];
  out[i] = f32_to_bf16(v);
}

// ---------------------------------------------------------------------------
// Generalized WMMA GEMM:  C[M,N] = A'[M,K] @ Bw.T   (Bw is bf16 [Npad,Kpad])
//   A' element (row,k):
//     gather mode (idxA!=0):  A[idxA[row]*K + k] + A2[idxB[row]*K + k]
//     concat mode (splitK>0): k<splitK ? A[row*splitK+k] : A2[row*(K-splitK)+k-splitK]
//     plain:                  A[row*K + k]
//   Output: idxOut ? atomicAdd(&C[idxOut[row]*N + n]) : C[row*N + n] = v
// Block: 128 threads (4 waves), 64x64 tile; wave = 32x32 (2x2 of 16x16 WMMA).
// B tile staged by global_load_async_to_lds_b128 (ASYNCcnt) overlapping the
// VALU A-stage (fp32->bf16 pack + fused gather); fenced by s_wait_asynccnt.
// ---------------------------------------------------------------------------
__global__ __launch_bounds__(128)
void wmma_gemm_kernel(const float* __restrict__ A,
                      const float* __restrict__ A2,
                      const int*   __restrict__ idxA,
                      const int*   __restrict__ idxB,
                      const unsigned short* __restrict__ Bw,
                      int ldB,
                      float*       __restrict__ C,
                      const int*   __restrict__ idxOut,
                      int M, int N, int K, int splitK)
{
  __shared__ unsigned int As[64][17];   // 64 rows x 16 bf16-pairs (+pad, 4B banks)
  __shared__ unsigned int Bs[64][20];   // 64 cols x 16 bf16-pairs; stride 80B:
                                        //   16B-aligned rows (async B128 writes)
                                        //   + conflict-free ds reads (gcd(20,64)=4)

  const int tid  = threadIdx.x;
  const int lane = tid & 31;
  const int wave = tid >> 5;
  const int wr   = (wave >> 1) * 32;
  const int wc   = (wave & 1) * 32;
  const int rowBase = blockIdx.y * 64;
  const int colBase = blockIdx.x * 64;

  v8f acc00 = {}; v8f acc01 = {}; v8f acc10 = {}; v8f acc11 = {};

  const int r16  = lane & 15;
  const int half = lane >> 4;

  for (int kb = 0; kb < K; kb += 32) {
    // ---- kick off async DMA of the bf16 B tile (64 cols x 32 k = 64B/row) ----
#pragma unroll
    for (int it = 0; it < 2; ++it) {
      int e = tid + it * 128;           // 256 x 16B transfers
      int c = e >> 2;                   // column within tile
      int q = e & 3;                    // 16B chunk within the 64B row
      const unsigned short* gp = Bw + (size_t)(colBase + c) * ldB + kb + q * 8;
      unsigned long long gaddr = (unsigned long long)(uintptr_t)gp;
      unsigned ldsoff = (unsigned)(uintptr_t)(&Bs[c][q * 4]);
      asm volatile("global_load_async_to_lds_b128 %0, %1, off"
                   :: "v"(ldsoff), "v"(gaddr) : "memory");
    }

    // ---- meanwhile: stage A tile (64 x 32) fp32 -> bf16 pairs via VALU ----
    for (int e = tid; e < 64 * 16; e += 128) {
      int r  = e >> 4;
      int kp = e & 15;
      int gm = rowBase + r;
      int k0 = kb + kp * 2;
      float f0 = 0.0f, f1 = 0.0f;
      if (gm < M) {
        if (idxA) {                                 // fused gather: h[src]+h0[etype]
          long ra = (long)idxA[gm] * K;
          long rb = (long)idxB[gm] * K;
          if (k0     < K) f0 = A[ra + k0]     + A2[rb + k0];
          if (k0 + 1 < K) f1 = A[ra + k0 + 1] + A2[rb + k0 + 1];
        } else if (splitK > 0) {                    // concat(cur, twin)
          int k = k0;
          if (k < K) f0 = (k < splitK) ? A[(long)gm*splitK + k]
                                       : A2[(long)gm*(K-splitK) + (k-splitK)];
          k = k0 + 1;
          if (k < K) f1 = (k < splitK) ? A[(long)gm*splitK + k]
                                       : A2[(long)gm*(K-splitK) + (k-splitK)];
        } else {
          if (k0     < K) f0 = A[(long)gm*K + k0];
          if (k0 + 1 < K) f1 = A[(long)gm*K + k0 + 1];
        }
      }
      As[r][kp] = pack_bf16x2(f0, f1);
    }

    // fence the async DMA, then workgroup barrier
    asm volatile("s_wait_asynccnt 0x0" ::: "memory");
    __syncthreads();

    // ---- fragment loads: contiguous k = 16*half + s, same order for A and B ----
    FragU a0, a1, b0, b1;
#pragma unroll
    for (int i = 0; i < 8; ++i) {
      // A 16x32 bf16: lanes 0-15 K=0-15 ; lanes 16-31 K=16-31
      int kkA = half * 8 + i;
      a0.u[i] = As[wr      + r16][kkA];
      a1.u[i] = As[wr + 16 + r16][kkA];
      // B 32x16 bf16: lanes 0-15 K=0-15 ; lanes 16-31 K=16-31
      int kkB = half * 8 + i;
      b0.u[i] = Bs[wc      + r16][kkB];
      b1.u[i] = Bs[wc + 16 + r16][kkB];
    }
    acc00 = __builtin_amdgcn_wmma_f32_16x16x32_bf16(false, a0.v, false, b0.v, (short)0, acc00, false, false);
    acc01 = __builtin_amdgcn_wmma_f32_16x16x32_bf16(false, a0.v, false, b1.v, (short)0, acc01, false, false);
    acc10 = __builtin_amdgcn_wmma_f32_16x16x32_bf16(false, a1.v, false, b0.v, (short)0, acc10, false, false);
    acc11 = __builtin_amdgcn_wmma_f32_16x16x32_bf16(false, a1.v, false, b1.v, (short)0, acc11, false, false);
    __syncthreads();
  }

  // ---- writeback: C/D layout lanes 0-15 M=g, lanes 16-31 M=8+g, N=lane&15 ----
  const int n16 = lane & 15;
  const int mo  = half * 8;
  auto storeTile = [&](v8f acc, int gmBase, int gnBase) {
#pragma unroll
    for (int g = 0; g < 8; ++g) {
      int gm = gmBase + mo + g;
      int gn = gnBase + n16;
      if (gm < M && gn < N) {
        float v = acc[g];
        if (idxOut) atomicAdd(&C[(long)idxOut[gm]*N + gn], v);
        else        C[(long)gm*N + gn] = v;
      }
    }
  };
  storeTile(acc00, rowBase + wr,      colBase + wc);
  storeTile(acc01, rowBase + wr,      colBase + wc + 16);
  storeTile(acc10, rowBase + wr + 16, colBase + wc);
  storeTile(acc11, rowBase + wr + 16, colBase + wc + 16);
}

// ---------------------------------------------------------------------------
// Support kernels
// ---------------------------------------------------------------------------
__global__ void k_zero(float* __restrict__ p, long n) {
  long i = (long)blockIdx.x * blockDim.x + threadIdx.x;
  if (i < n) p[i] = 0.0f;
}

__global__ __launch_bounds__(256)
void k_l2norm_rows(const float* __restrict__ in, float* __restrict__ out, int H) {
  __shared__ float red[256];
  int r = blockIdx.x, j = threadIdx.x;
  float v = (j < H) ? in[(long)r * H + j] : 0.0f;
  red[j] = v * v;
  __syncthreads();
  for (int s = 128; s > 0; s >>= 1) { if (j < s) red[j] += red[j + s]; __syncthreads(); }
  float nrm = fmaxf(sqrtf(red[0]), 1e-12f);
  if (j < H) out[(long)r * H + j] = v / nrm;
}

__global__ void k_rel_accum(const float* __restrict__ h, const int* __restrict__ e_ids,
                            const int* __restrict__ r_ids, float* __restrict__ sums, long total) {
  long i = (long)blockIdx.x * blockDim.x + threadIdx.x;
  if (i >= total) return;
  int m = (int)(i / HD), j = (int)(i % HD);
  atomicAdd(&sums[(long)r_ids[m] * HD + j], h[(long)e_ids[m] * HD + j]);
}

__global__ void k_cnt_accum(const int* __restrict__ ids, float* __restrict__ cnt, int n) {
  int i = blockIdx.x * blockDim.x + threadIdx.x;
  if (i < n) atomicAdd(&cnt[ids[i]], 1.0f);
}

__global__ void k_xinput(const float* __restrict__ emb_rel, const float* __restrict__ sums,
                         const float* __restrict__ cnt, float* __restrict__ x, long total) {
  long i = (long)blockIdx.x * blockDim.x + threadIdx.x;
  if (i >= total) return;
  int r = (int)(i / (2 * HD)), j = (int)(i % (2 * HD));
  x[i] = (j < HD) ? emb_rel[(long)r * HD + j]
                  : sums[(long)r * HD + (j - HD)] / fmaxf(cnt[r], 1.0f);
}

__global__ __launch_bounds__(256)
void k_gru(const float* __restrict__ gi, const float* __restrict__ gh,
           const float* __restrict__ bi, const float* __restrict__ bh,
           float* __restrict__ h0) {
  __shared__ float red[256];
  int r = blockIdx.x, j = threadIdx.x;
  float val = 0.0f;
  if (j < HD) {
    long b = (long)r * 3 * HD;
    float ir = gi[b + j]          + bi[j];
    float iz = gi[b + HD + j]     + bi[HD + j];
    float in = gi[b + 2 * HD + j] + bi[2 * HD + j];
    float hr = gh[b + j]          + bh[j];
    float hz = gh[b + HD + j]     + bh[HD + j];
    float hn = gh[b + 2 * HD + j] + bh[2 * HD + j];
    float rg = sigmoidf_(ir + hr);
    float z  = sigmoidf_(iz + hz);
    float nn = tanhf(in + rg * hn);
    val = (1.0f - z) * nn + z * h0[(long)r * HD + j];
  }
  red[j] = val * val;
  __syncthreads();
  for (int s = 128; s > 0; s >>= 1) { if (j < s) red[j] += red[j + s]; __syncthreads(); }
  float nrm = fmaxf(sqrtf(red[0]), 1e-12f);
  if (j < HD) h0[(long)r * HD + j] = val / nrm;
}

__global__ void k_combine(const float* __restrict__ agg, const float* __restrict__ loopW,
                          const float* __restrict__ evolveW, const float* __restrict__ indeg,
                          float* __restrict__ out, long total) {
  long i = (long)blockIdx.x * blockDim.x + threadIdx.x;
  if (i >= total) return;
  int n = (int)(i / HD);
  float d = indeg[n];
  float v = agg[i] / fmaxf(d, 1.0f) + (d > 0.0f ? loopW[i] : evolveW[i]);
  out[i] = (v >= 0.0f) ? v : RRELU_SLOPE * v;
}

__global__ __launch_bounds__(256)
void k_gate(float* __restrict__ cur, const float* __restrict__ xg,
            const float* __restrict__ fc_b, const float* __restrict__ proj_w,
            const float* __restrict__ proj_b, const float* __restrict__ twin) {
  __shared__ float red[256];
  int r = blockIdx.x, j = threadIdx.x;
  float acc = 0.0f;
  if (j < HD) acc = tanhf(xg[(long)r * HD + j] + fc_b[j]) * proj_w[j];
  red[j] = acc;
  __syncthreads();
  for (int s = 128; s > 0; s >>= 1) { if (j < s) red[j] += red[j + s]; __syncthreads(); }
  float g = sigmoidf_(red[0] + proj_b[0]);
  if (j < HD) {
    long idx = (long)r * HD + j;
    cur[idx] = (1.0f - g) * cur[idx] + g * twin[idx];
  }
}

__global__ void k_timegate(float* __restrict__ h, const float* __restrict__ timeW,
                           const float* __restrict__ time_b, const float* __restrict__ cur,
                           float* __restrict__ out_t, long total) {
  long i = (long)blockIdx.x * blockDim.x + threadIdx.x;
  if (i >= total) return;
  int j = (int)(i % HD);
  float tw = sigmoidf_(timeW[i] + time_b[j]);
  float hn = tw * cur[i] + (1.0f - tw) * h[i];
  h[i] = hn;
  out_t[i] = hn;
}

// ---------------------------------------------------------------------------
// Host orchestration
// ---------------------------------------------------------------------------
static inline void launch_gemm(const float* A, const float* A2, const int* idxA, const int* idxB,
                               const unsigned short* Bw, int ldB, float* C, const int* idxOut,
                               int M, int N, int K, int splitK, hipStream_t s) {
  dim3 grid((N + 63) / 64, (M + 63) / 64);
  wmma_gemm_kernel<<<grid, 128, 0, s>>>(A, A2, idxA, idxB, Bw, ldB, C, idxOut, M, N, K, splitK);
}
static inline void launch_zero(float* p, long n, hipStream_t s) {
  k_zero<<<(unsigned)((n + 255) / 256), 256, 0, s>>>(p, n);
}
static inline void launch_conv(const float* W, unsigned short* out, int N, int K,
                               int Npad, int Kpad, int trans, hipStream_t s) {
  long total = (long)Npad * Kpad;
  k_conv_w<<<(unsigned)((total + 255) / 256), 256, 0, s>>>(W, out, N, K, Kpad, total, trans);
}

extern "C" void kernel_launch(void* const* d_in, const int* in_sizes, int n_in,
                              void* d_out, int out_size, void* d_ws, size_t ws_size,
                              hipStream_t stream) {
  const float* dynamic_emb = (const float*)d_in[0];
  const float* emb_rel     = (const float*)d_in[1];
  const float* gru_wi      = (const float*)d_in[2];   // [3H, 2H]
  const float* gru_wh      = (const float*)d_in[3];   // [3H, H]
  const float* gru_bi      = (const float*)d_in[4];
  const float* gru_bh      = (const float*)d_in[5];
  const float* W_neigh     = (const float*)d_in[6];   // [2,H,H]
  const float* W_loop      = (const float*)d_in[7];
  const float* W_evolve    = (const float*)d_in[8];
  const float* fc_w        = (const float*)d_in[9];   // [H, 2H]
  const float* fc_b        = (const float*)d_in[10];
  const float* proj_w      = (const float*)d_in[11];  // [1, H]
  const float* proj_b      = (const float*)d_in[12];
  const float* time_w      = (const float*)d_in[13];  // [H, H]
  const float* time_b      = (const float*)d_in[14];
  const float* twin        = (const float*)d_in[15];  // [T, N, H]
  const int*   src         = (const int*)d_in[16];
  const int*   dst         = (const int*)d_in[17];
  const int*   etype       = (const int*)d_in[18];
  const int*   e_ids       = (const int*)d_in[19];
  const int*   r_ids       = (const int*)d_in[20];
  (void)in_sizes; (void)n_in; (void)out_size; (void)ws_size;

  // ---- fp32 workspace layout ----
  float* ws = (float*)d_ws;
  size_t NH = (size_t)NENT * HD;
  float* h       = ws;                 // [N,H]  entity carry
  float* bufX    = h       + NH;
  float* bufY    = bufX    + NH;
  float* loopW   = bufY    + NH;       // reused for xg
  float* evolveW = loopW   + NH;       // reused for timeW
  float* agg     = evolveW + NH;
  float* h0      = agg     + NH;                       // [R2,H]
  float* sums    = h0      + (size_t)R2 * HD;
  float* cnt     = sums    + (size_t)R2 * HD;
  float* x_input = cnt     + R2;                       // [R2,2H]
  float* gi      = x_input + (size_t)R2 * 2 * HD;      // [R2,3H]
  float* gh      = gi      + (size_t)R2 * 3 * HD;
  float* indeg   = gh      + (size_t)R2 * 3 * HD;      // [N]

  // ---- bf16 padded weight pool (after fp32 region, 16B aligned) ----
  unsigned short* wpool = (unsigned short*)(ws + 49040512);
  unsigned short* wiB   = wpool;                 // [640,416]
  unsigned short* whB   = wiB  + 640 * 416;      // [640,224]
  unsigned short* WnB   = whB  + 640 * 224;      // 2 x [256,224]
  unsigned short* WlB   = WnB  + 2 * 256 * 224;
  unsigned short* WeB   = WlB  + 2 * 256 * 224;
  unsigned short* fcB   = WeB  + 2 * 256 * 224;  // [256,416]
  unsigned short* timeB = fcB  + 256 * 416;      // [256,224]

  // ---- one-time per-launch weight conversion to padded bf16 [N,K] ----
  launch_conv(gru_wi, wiB, 600, 400, 640, 416, 0, stream);
  launch_conv(gru_wh, whB, 600, 200, 640, 224, 0, stream);
  for (int l = 0; l < 2; ++l) {
    launch_conv(W_neigh  + (size_t)l * HD * HD, WnB + (size_t)l * 256 * 224, HD, HD, 256, 224, 1, stream);
    launch_conv(W_loop   + (size_t)l * HD * HD, WlB + (size_t)l * 256 * 224, HD, HD, 256, 224, 1, stream);
    launch_conv(W_evolve + (size_t)l * HD * HD, WeB + (size_t)l * 256 * 224, HD, HD, 256, 224, 1, stream);
  }
  launch_conv(fc_w,   fcB,   HD, 400, 256, 416, 0, stream);
  launch_conv(time_w, timeB, HD, HD,  256, 224, 1, stream);

  // h = l2norm(dynamic_emb); h0 = emb_rel
  k_l2norm_rows<<<NENT, 256, 0, stream>>>(dynamic_emb, h, HD);
  hipMemcpyAsync(h0, emb_rel, (size_t)R2 * HD * sizeof(float),
                 hipMemcpyDeviceToDevice, stream);

  for (int t = 0; t < TT; ++t) {
    const int*   src_t   = src   + (size_t)t * EE;
    const int*   dst_t   = dst   + (size_t)t * EE;
    const int*   etype_t = etype + (size_t)t * EE;
    const int*   e_t     = e_ids + (size_t)t * MM;
    const int*   r_t     = r_ids + (size_t)t * MM;
    const float* twin_t  = twin  + (size_t)t * NH;
    float*       out_t   = (float*)d_out + (size_t)t * NH;

    // ---- relation means ----
    launch_zero(sums, (long)R2 * HD, stream);
    launch_zero(cnt, R2, stream);
    launch_zero(indeg, NENT, stream);
    k_rel_accum<<<(unsigned)(((long)MM * HD + 255) / 256), 256, 0, stream>>>(h, e_t, r_t, sums, (long)MM * HD);
    k_cnt_accum<<<(MM + 255) / 256, 256, 0, stream>>>(r_t, cnt, MM);
    k_xinput<<<(unsigned)(((long)R2 * 2 * HD + 255) / 256), 256, 0, stream>>>(emb_rel, sums, cnt, x_input, (long)R2 * 2 * HD);

    // ---- GRU over relations (WMMA GEMMs + gates) ----
    launch_gemm(x_input, nullptr, nullptr, nullptr, wiB, 416, gi, nullptr,
                R2, 3 * HD, 2 * HD, 0, stream);                 // gi = x @ wi.T
    launch_gemm(h0, nullptr, nullptr, nullptr, whB, 224, gh, nullptr,
                R2, 3 * HD, HD, 0, stream);                     // gh = h0 @ wh.T
    k_gru<<<R2, 256, 0, stream>>>(gi, gh, gru_bi, gru_bh, h0);

    // ---- in-degree ----
    k_cnt_accum<<<(EE + 255) / 256, 256, 0, stream>>>(dst_t, indeg, EE);

    // ---- two UnionRGCN layers ----
    const float* lin = h;
    float* louts[2] = { bufX, bufY };
    for (int l = 0; l < 2; ++l) {
      const unsigned short* WnBl = WnB + (size_t)l * 256 * 224;
      const unsigned short* WlBl = WlB + (size_t)l * 256 * 224;
      const unsigned short* WeBl = WeB + (size_t)l * 256 * 224;
      launch_gemm(lin, nullptr, nullptr, nullptr, WlBl, 224, loopW,   nullptr, NENT, HD, HD, 0, stream);
      launch_gemm(lin, nullptr, nullptr, nullptr, WeBl, 224, evolveW, nullptr, NENT, HD, HD, 0, stream);
      launch_zero(agg, (long)NH, stream);
      // fused gather (h[src]+h0[etype]) @ W_neigh with atomic scatter to agg[dst]
      launch_gemm(lin, h0, src_t, etype_t, WnBl, 224, agg, dst_t, EE, HD, HD, 0, stream);
      k_combine<<<(unsigned)((NH + 255) / 256), 256, 0, stream>>>(agg, loopW, evolveW, indeg, louts[l], (long)NH);
      lin = louts[l];
    }

    // ---- cur = l2norm(hn) (in place in bufY) ----
    k_l2norm_rows<<<NENT, 256, 0, stream>>>(bufY, bufY, HD);

    // ---- twin fusion gate: xg = concat(cur,twin) @ fc_w.T (reuse loopW) ----
    launch_gemm(bufY, twin_t, nullptr, nullptr, fcB, 416, loopW, nullptr,
                NENT, HD, 2 * HD, /*splitK=*/HD, stream);
    k_gate<<<NENT, 256, 0, stream>>>(bufY, loopW, fc_b, proj_w, proj_b, twin_t);

    // ---- entity time gate: timeW = h @ time_w (reuse evolveW) ----
    launch_gemm(h, nullptr, nullptr, nullptr, timeB, 224, evolveW, nullptr,
                NENT, HD, HD, 0, stream);
    k_timegate<<<(unsigned)((NH + 255) / 256), 256, 0, stream>>>(h, evolveW, time_b, bufY, out_t, (long)NH);
  }
}
